// Attention_2104533975762
// MI455X (gfx1250) — matrix operand order repaired and checked by reading, not compiled, not run on hardware
//
#include <hip/hip_runtime.h>
#include <hip/hip_bf16.h>

typedef __bf16 bf16;
typedef __bf16 v16bf __attribute__((ext_vector_type(16)));
typedef __bf16 v8bf  __attribute__((ext_vector_type(8)));
typedef __bf16 v4bf  __attribute__((ext_vector_type(4)));
typedef float  v8f   __attribute__((ext_vector_type(8)));

#define DIM      1024
#define HEADS    16
#define HEAD_DIM 64
#define SEQ      2048
#define BATCH    4
#define TOKENS   (BATCH * SEQ)
#define QKV_DIM  (3 * DIM)
#define SCALE    0.125f

// ---------------------------------------------------------------------------
// CDNA5 async memory->LDS copy (ASYNCcnt-tracked, bypasses VGPRs)
// GVS mode: saddr = uniform 64-bit base, vaddr = per-lane 32-bit byte offset,
// IOFFSET applies to both LDS and global addresses.
// ---------------------------------------------------------------------------
#define ASYNC_B128(ldsv, offv, base, IMM)                                  \
  asm volatile("global_load_async_to_lds_b128 %0, %1, %2 offset:" #IMM     \
               :: "v"(ldsv), "v"(offv), "s"(base) : "memory")

__device__ __forceinline__ void async_copy64B(void* lds, const void* gbase,
                                              unsigned byteOff) {
  unsigned l = (unsigned)(uintptr_t)lds;  // low 32 bits of generic = LDS offset
  ASYNC_B128(l, byteOff, gbase, 0);
  ASYNC_B128(l, byteOff, gbase, 16);
  ASYNC_B128(l, byteOff, gbase, 32);
  ASYNC_B128(l, byteOff, gbase, 48);
}

#define WAIT_ASYNC() asm volatile("s_wait_asynccnt 0x0" ::: "memory")

// ---------------------------------------------------------------------------
// helpers
// ---------------------------------------------------------------------------
__device__ __forceinline__ v16bf cat16(v8bf lo, v8bf hi) {
  v16bf r;
#pragma unroll
  for (int i = 0; i < 8; ++i) { r[i] = lo[i]; r[8 + i] = hi[i]; }
  return r;
}

__device__ __forceinline__ float hmax16(float v) {
#pragma unroll
  for (int m = 8; m >= 1; m >>= 1) v = fmaxf(v, __shfl_xor(v, m, 32));
  return v;
}
__device__ __forceinline__ float hsum16(float v) {
#pragma unroll
  for (int m = 8; m >= 1; m >>= 1) v += __shfl_xor(v, m, 32);
  return v;
}

__device__ __forceinline__ v8f wmma_bf16(v16bf a, v16bf b, v8f c) {
  return __builtin_amdgcn_wmma_f32_16x16x32_bf16(false, a, false, b,
                                                 (short)0, c, false, false);
}

// ---------------------------------------------------------------------------
// fp32 -> bf16 conversion
// ---------------------------------------------------------------------------
__global__ __launch_bounds__(256) void cvt_f32_bf16(const float* __restrict__ in,
                                                    bf16* __restrict__ out, int n) {
  int i = (blockIdx.x * 256 + threadIdx.x) * 4;
  if (i + 3 < n) {
    float4 v = *(const float4*)(in + i);
    v4bf o;
    o[0] = (bf16)v.x; o[1] = (bf16)v.y; o[2] = (bf16)v.z; o[3] = (bf16)v.w;
    *(v4bf*)(out + i) = o;
  } else {
    for (; i < n; ++i) out[i] = (bf16)in[i];
  }
}

// ---------------------------------------------------------------------------
// shared 128x128 GEMM mainloop:  acc = A[M,K] * W[N,K]^T  (row-major bf16)
// double-buffered LDS, async global->LDS staging overlapped with WMMA.
// 8 waves in a 2x4 grid; each wave owns 64x32 (4x2 WMMA tiles).
// ---------------------------------------------------------------------------
__device__ __forceinline__ void gemm_tile_128x128(
    const bf16* __restrict__ A, const bf16* __restrict__ W, int K,
    int mBase, int nBase, bf16 (*sA)[128][72], bf16 (*sB)[128][72],
    v8f acc[4][2]) {
  const int tid  = threadIdx.x;
  const int lane = tid & 31, wave = tid >> 5;
  const int half = lane >> 4, l16 = lane & 15;
  const int wm = wave >> 2, wn = wave & 3;
  const int row = tid >> 1, ch = tid & 1;

#pragma unroll
  for (int mi = 0; mi < 4; ++mi)
#pragma unroll
    for (int ni = 0; ni < 2; ++ni)
#pragma unroll
      for (int e = 0; e < 8; ++e) acc[mi][ni][e] = 0.f;

  const bf16* gA = A + (size_t)mBase * K;
  const bf16* gB = W + (size_t)nBase * K;
  const unsigned thrOff = (unsigned)(row * K + ch * 32) * 2u;  // bytes

  // prologue: stage kb = 0 into buffer 0
  async_copy64B(&sA[0][row][ch * 32], gA, thrOff);
  async_copy64B(&sB[0][row][ch * 32], gB, thrOff);

  int buf = 0;
  for (int kb = 0; kb < K; kb += 64) {
    WAIT_ASYNC();
    __syncthreads();
    if (kb + 64 < K) {  // issue next stage; overlaps with compute below
      async_copy64B(&sA[buf ^ 1][row][ch * 32], gA, thrOff + (kb + 64) * 2);
      async_copy64B(&sB[buf ^ 1][row][ch * 32], gB, thrOff + (kb + 64) * 2);
    }
#pragma unroll
    for (int ks = 0; ks < 64; ks += 32) {
      v16bf aF[4], bF[2];
#pragma unroll
      for (int mi = 0; mi < 4; ++mi) {           // A-layout: two 16B chunks
        const bf16* r = &sA[buf][wm * 64 + mi * 16 + l16][0];
        aF[mi] = cat16(*(const v8bf*)(r + ks + 16 * half),
                       *(const v8bf*)(r + ks + 16 * half + 8));
      }
#pragma unroll
      for (int ni = 0; ni < 2; ++ni) {           // B-layout: 32B contiguous K
        const bf16* r = &sB[buf][wn * 32 + ni * 16 + l16][0];
        const v8bf* p = (const v8bf*)(r + ks + 16 * half);
        bF[ni] = cat16(p[0], p[1]);
      }
#pragma unroll
      for (int mi = 0; mi < 4; ++mi)
#pragma unroll
        for (int ni = 0; ni < 2; ++ni)
          acc[mi][ni] = wmma_bf16(aF[mi], bF[ni], acc[mi][ni]);
    }
    buf ^= 1;
  }
}

// ---------------------------------------------------------------------------
// QKV GEMM: xb[T,1024] * Wqkv[3072,1024]^T
//   Q, K scatter to [B*H, SEQ, 64]; V scatters TRANSPOSED to [B*H, 64, SEQ]
//   so the attention kernel can async-copy V tiles with no LDS transpose.
// ---------------------------------------------------------------------------
__global__ __launch_bounds__(256) void qkv_gemm_kernel(
    const bf16* __restrict__ xb, const bf16* __restrict__ Wq,
    bf16* __restrict__ Qb, bf16* __restrict__ Kb, bf16* __restrict__ Vb) {
  __shared__ bf16 sA[2][128][72];
  __shared__ bf16 sB[2][128][72];
  v8f acc[4][2];
  const int mBase = blockIdx.y * 128;
  const int nBase = blockIdx.x * 128;
  gemm_tile_128x128(xb, Wq, DIM, mBase, nBase, sA, sB, acc);

  const int lane = threadIdx.x & 31, wave = threadIdx.x >> 5;
  const int half = lane >> 4, l16 = lane & 15;
  const int wm = wave >> 2, wn = wave & 3;
#pragma unroll
  for (int mi = 0; mi < 4; ++mi)
#pragma unroll
    for (int ni = 0; ni < 2; ++ni) {
      int col = nBase + wn * 32 + ni * 16 + l16;  // [0, 3072)
      int t = col >> 10, c = col & 1023;
      int h = c >> 6, d = c & 63;
#pragma unroll
      for (int r = 0; r < 8; ++r) {
        int tok = mBase + wm * 64 + mi * 16 + r + 8 * half;  // [0, 8192)
        int b = tok >> 11, n = tok & 2047;
        int bh = b * HEADS + h;
        bf16 v = (bf16)acc[mi][ni][r];
        if (t == 2) {
          Vb[((size_t)bh * HEAD_DIM + d) * SEQ + n] = v;   // transposed
        } else {
          bf16* dst = (t == 0) ? Qb : Kb;
          dst[((size_t)bh * SEQ + n) * HEAD_DIM + d] = v;
        }
      }
    }
}

// ---------------------------------------------------------------------------
// Flash attention: 1 block per (b*h, 128-query tile); 8 waves x 16 rows each.
// K and V tiles double-buffered via async global->LDS copies.
// ---------------------------------------------------------------------------
__global__ __launch_bounds__(256) void attn_kernel(
    const bf16* __restrict__ Qb, const bf16* __restrict__ Kb,
    const bf16* __restrict__ Vb, bf16* __restrict__ Ob) {
  __shared__ bf16 sK[2][128][72];    // [key][d]  (= B-operand layout for S)
  __shared__ bf16 sVt[2][64][136];   // [d][key]  (= B-operand layout for O)
  __shared__ bf16 sP[8][16][136];    // per-wave P, C-layout -> A-layout bridge

  const int tid  = threadIdx.x;
  const int lane = tid & 31, wave = tid >> 5;
  const int half = lane >> 4, l16 = lane & 15;
  const int bh    = blockIdx.y;         // b*HEADS + h
  const int qBase = blockIdx.x * 128;   // sequence offset
  const int b = bh >> 4, h = bh & 15;

  const bf16* Kbase = Kb + (size_t)bh * SEQ * HEAD_DIM;
  const bf16* Vbase = Vb + (size_t)bh * HEAD_DIM * SEQ;  // [64][2048]

  // staging roles
  const int rowK = tid >> 1, chK = tid & 1;  // K tile: 128 rows x 64 (2 thr/row)
  const int rowV = tid >> 2, qV = tid & 3;   // V tile: 64 rows x 128 (4 thr/row)

  // Q A-fragments loaded once, straight from global (two b128 per K-step)
  const bf16* qrow =
      Qb + ((size_t)bh * SEQ + qBase + wave * 16 + l16) * HEAD_DIM;
  v16bf aQ[2];
#pragma unroll
  for (int ks = 0; ks < 2; ++ks)
    aQ[ks] = cat16(*(const v8bf*)(qrow + ks * 32 + 16 * half),
                   *(const v8bf*)(qrow + ks * 32 + 16 * half + 8));

  float mRun[8], lRun[8], alpha[8];
  v8f accO[4];
#pragma unroll
  for (int r = 0; r < 8; ++r) { mRun[r] = -1e30f; lRun[r] = 0.f; }
#pragma unroll
  for (int jd = 0; jd < 4; ++jd)
#pragma unroll
    for (int e = 0; e < 8; ++e) accO[jd][e] = 0.f;

  // prologue: stage key-tile 0 into buffer 0
  async_copy64B(&sK[0][rowK][chK * 32], Kbase,
                (unsigned)(rowK * HEAD_DIM + chK * 32) * 2u);
  async_copy64B(&sVt[0][rowV][qV * 32], Vbase,
                (unsigned)(rowV * SEQ + qV * 32) * 2u);

  int buf = 0;
  for (int kt = 0; kt < SEQ; kt += 128) {
    WAIT_ASYNC();
    __syncthreads();
    if (kt + 128 < SEQ) {  // overlap next K/V tile fetch with this tile's math
      async_copy64B(&sK[buf ^ 1][rowK][chK * 32], Kbase,
                    (unsigned)((kt + 128 + rowK) * HEAD_DIM + chK * 32) * 2u);
      async_copy64B(&sVt[buf ^ 1][rowV][qV * 32], Vbase,
                    (unsigned)(rowV * SEQ + kt + 128 + qV * 32) * 2u);
    }

    // S = (Q K^T) * SCALE : 8 key sub-tiles x 2 K-steps = 16 WMMA
    v8f accS[8];
#pragma unroll
    for (int j = 0; j < 8; ++j) {
#pragma unroll
      for (int e = 0; e < 8; ++e) accS[j][e] = 0.f;
      const bf16* kr = &sK[buf][j * 16 + l16][0];
#pragma unroll
      for (int ks = 0; ks < 2; ++ks) {
        const v8bf* p = (const v8bf*)(kr + ks * 32 + 16 * half);
        accS[j] = wmma_bf16(aQ[ks], cat16(p[0], p[1]), accS[j]);
      }
#pragma unroll
      for (int e = 0; e < 8; ++e) accS[j][e] *= SCALE;
    }

    // online softmax (C-layout rows live within one 16-lane half)
#pragma unroll
    for (int r = 0; r < 8; ++r) {
      float mx = accS[0][r];
#pragma unroll
      for (int j = 1; j < 8; ++j) mx = fmaxf(mx, accS[j][r]);
      mx = hmax16(mx);
      float nm = fmaxf(mRun[r], mx);
      alpha[r] = __expf(mRun[r] - nm);
      mRun[r]  = nm;
      float rs = 0.f;
#pragma unroll
      for (int j = 0; j < 8; ++j) {
        float p = __expf(accS[j][r] - nm);
        accS[j][r] = p;
        rs += p;
      }
      rs = hsum16(rs);
      lRun[r] = lRun[r] * alpha[r] + rs;
    }
#pragma unroll
    for (int jd = 0; jd < 4; ++jd)
#pragma unroll
      for (int r = 0; r < 8; ++r) accO[jd][r] *= alpha[r];

    // P: C-layout -> A-layout via per-wave LDS
#pragma unroll
    for (int j = 0; j < 8; ++j)
#pragma unroll
      for (int r = 0; r < 8; ++r)
        sP[wave][r + 8 * half][j * 16 + l16] = (bf16)accS[j][r];
    asm volatile("s_wait_dscnt 0x0" ::: "memory");

    // O += P @ V : 4 K-steps x 4 d-tiles = 16 WMMA
#pragma unroll
    for (int ks = 0; ks < 4; ++ks) {
      const bf16* pr = &sP[wave][l16][0];
      v16bf aP = cat16(*(const v8bf*)(pr + ks * 32 + 16 * half),
                       *(const v8bf*)(pr + ks * 32 + 16 * half + 8));
#pragma unroll
      for (int jd = 0; jd < 4; ++jd) {
        const bf16* vr = &sVt[buf][jd * 16 + l16][0];
        const v8bf* p  = (const v8bf*)(vr + ks * 32 + 16 * half);
        accO[jd] = wmma_bf16(aP, cat16(p[0], p[1]), accO[jd]);
      }
    }
    buf ^= 1;
  }

  // epilogue: O /= L, store to [token][h*64+d] bf16
#pragma unroll
  for (int jd = 0; jd < 4; ++jd) {
    int d = jd * 16 + l16;
#pragma unroll
    for (int r = 0; r < 8; ++r) {
      int n = qBase + wave * 16 + r + 8 * half;
      float o = accO[jd][r] / lRun[r];
      Ob[(size_t)(b * SEQ + n) * DIM + h * HEAD_DIM + d] = (bf16)o;
    }
  }
}

// ---------------------------------------------------------------------------
// proj GEMM: Ob[T,1024] * Wp[1024,1024]^T + bias -> fp32 out
// ---------------------------------------------------------------------------
__global__ __launch_bounds__(256) void proj_gemm_kernel(
    const bf16* __restrict__ Ob, const bf16* __restrict__ Wp,
    const float* __restrict__ bias, float* __restrict__ out) {
  __shared__ bf16 sA[2][128][72];
  __shared__ bf16 sB[2][128][72];
  v8f acc[4][2];
  const int mBase = blockIdx.y * 128;
  const int nBase = blockIdx.x * 128;
  gemm_tile_128x128(Ob, Wp, DIM, mBase, nBase, sA, sB, acc);

  const int lane = threadIdx.x & 31, wave = threadIdx.x >> 5;
  const int half = lane >> 4, l16 = lane & 15;
  const int wm = wave >> 2, wn = wave & 3;
#pragma unroll
  for (int mi = 0; mi < 4; ++mi)
#pragma unroll
    for (int ni = 0; ni < 2; ++ni) {
      int col = nBase + wn * 32 + ni * 16 + l16;
      float bv = bias[col];
#pragma unroll
      for (int r = 0; r < 8; ++r) {
        int tok = mBase + wm * 64 + mi * 16 + r + 8 * half;
        out[(size_t)tok * DIM + col] = acc[mi][ni][r] + bv;
      }
    }
}

// ---------------------------------------------------------------------------
extern "C" void kernel_launch(void* const* d_in, const int* in_sizes, int n_in,
                              void* d_out, int out_size, void* d_ws,
                              size_t ws_size, hipStream_t stream) {
  const float* x      = (const float*)d_in[0];
  const float* W_qkv  = (const float*)d_in[1];
  const float* W_proj = (const float*)d_in[2];
  const float* b_proj = (const float*)d_in[3];
  float* out = (float*)d_out;

  char* ws = (char*)d_ws;
  size_t off = 0;
  auto alloc = [&](size_t bytes) {
    char* p = ws + off;
    off += (bytes + 255) & ~(size_t)255;
    return p;
  };
  bf16* xb  = (bf16*)alloc((size_t)TOKENS * DIM * 2);
  bf16* Wqb = (bf16*)alloc((size_t)QKV_DIM * DIM * 2);
  bf16* Wpb = (bf16*)alloc((size_t)DIM * DIM * 2);
  bf16* Qb  = (bf16*)alloc((size_t)TOKENS * DIM * 2);
  bf16* Kb  = (bf16*)alloc((size_t)TOKENS * DIM * 2);
  bf16* Vb  = (bf16*)alloc((size_t)TOKENS * DIM * 2);   // [B*H, 64, SEQ]
  bf16* Ob  = (bf16*)alloc((size_t)TOKENS * DIM * 2);

  int nx = TOKENS * DIM, nq = QKV_DIM * DIM, np = DIM * DIM;
  cvt_f32_bf16<<<nx / 4 / 256, 256, 0, stream>>>(x, xb, nx);
  cvt_f32_bf16<<<nq / 4 / 256, 256, 0, stream>>>(W_qkv, Wqb, nq);
  cvt_f32_bf16<<<np / 4 / 256, 256, 0, stream>>>(W_proj, Wpb, np);

  qkv_gemm_kernel<<<dim3(QKV_DIM / 128, TOKENS / 128), 256, 0, stream>>>(
      xb, Wqb, Qb, Kb, Vb);
  attn_kernel<<<dim3(SEQ / 128, BATCH * HEADS), 256, 0, stream>>>(Qb, Kb, Vb,
                                                                  Ob);
  proj_gemm_kernel<<<dim3(DIM / 128, TOKENS / 128), 256, 0, stream>>>(
      Ob, Wpb, b_proj, out);
}
